// KLinear_6356551598116
// MI455X (gfx1250) — matrix operand order repaired and checked by reading, not compiled, not run on hardware
//
#include <hip/hip_runtime.h>
#include <hip/hip_bf16.h>
#include <stdint.h>

typedef __attribute__((ext_vector_type(16))) __bf16 v16bf;
typedef __attribute__((ext_vector_type(8)))  float  v8f;

#define K_TOTAL 4096
#define N_TOTAL 4096
#define M_TOTAL 16384
#define BM 128
#define BN 256
#define BK 64
#define LDA 72   // halves per LDS row (144B: 16B-aligned, bank-conflict-breaking)
#define LDB 72

__device__ __forceinline__ unsigned short f2bf(float f) {
  uint32_t u = __float_as_uint(f);
  u += 0x7FFFu + ((u >> 16) & 1u);   // round-to-nearest-even
  return (unsigned short)(u >> 16);
}

// Async copy 16B from global directly into LDS (ASYNCcnt-tracked, no VGPR data path).
__device__ __forceinline__ void async_copy_b128(uint32_t lds_byte_off, const void* gptr) {
  asm volatile("global_load_async_to_lds_b128 %0, %1, off"
               :
               : "v"(lds_byte_off), "v"(gptr)
               : "memory");
}

__device__ __forceinline__ void wait_async_zero() {
  asm volatile("s_wait_asynccnt 0x0" ::: "memory");
}

// Phase 1: Wt[j (OUT)][i (IN)] = (kron(L0,R0)+kron(L1,R1)+kron(L2,R2))/3, bf16, transposed.
__global__ __launch_bounds__(256)
void KLinear_build_wt(const float* __restrict__ L0, const float* __restrict__ R0,
                      const float* __restrict__ L1, const float* __restrict__ R1,
                      const float* __restrict__ L2, const float* __restrict__ R2,
                      unsigned short* __restrict__ Wt) {
  uint32_t idx = blockIdx.x * 256u + threadIdx.x;     // 16M elements
  uint32_t i = idx & 4095u;                           // IN index (K), contiguous
  uint32_t j = idx >> 12;                             // OUT index (N)
  float v = L0[(i >> 6) * 64u  + (j >> 6)] * R0[(i & 63u)  * 64u  + (j & 63u)]
          + L1[(i >> 7) * 32u  + (j >> 7)] * R1[(i & 127u) * 128u + (j & 127u)]
          + L2[(i >> 5) * 128u + (j >> 5)] * R2[(i & 31u)  * 32u  + (j & 31u)];
  Wt[(uint64_t)j * 4096u + i] = f2bf(v * (1.0f / 3.0f));
}

// Phase 2: out = x @ W + bias. 8 waves, 64x64 per wave, 128x256 block tile,
// double-buffered LDS; B tile staged with global_load_async_to_lds_b128.
__global__ __launch_bounds__(256)
void KLinear_gemm(const float* __restrict__ X, const unsigned short* __restrict__ Wt,
                  const float* __restrict__ bias, float* __restrict__ Out) {
  __shared__ unsigned short sA[2][BM * LDA];   // [m][k] bf16
  __shared__ unsigned short sB[2][BN * LDB];   // [n][k] bf16 (Wt pre-transposed)

  const uint32_t tid   = threadIdx.x;
  const uint32_t lane  = tid & 31u;
  const uint32_t wave  = tid >> 5;          // 8 waves
  const uint32_t wm    = wave >> 2;         // 0..1  -> 64-row band
  const uint32_t wn    = wave & 3u;         // 0..3  -> 64-col band
  const uint32_t lhalf = lane >> 4;         // 0/1
  const uint32_t l15   = lane & 15u;
  const uint32_t m0    = blockIdx.y * BM;
  const uint32_t n0    = blockIdx.x * BN;

  v8f acc[4][4];
#pragma unroll
  for (int t = 0; t < 4; ++t)
#pragma unroll
    for (int u = 0; u < 4; ++u)
      acc[t][u] = (v8f){0.f, 0.f, 0.f, 0.f, 0.f, 0.f, 0.f, 0.f};

  // Stage one 128x64 A tile (fp32->bf16, via VGPRs) and one 256x64 B tile
  // (async global->LDS, ASYNCcnt) into LDS buffer `buf`.
  auto stage = [&](uint32_t kk, int buf) {
    // B tile: 2048 x 16B chunks, 8 per thread, straight to LDS.
#pragma unroll
    for (int p = 0; p < 8; ++p) {
      uint32_t q = tid + (uint32_t)p * 256u;      // 0..2047
      uint32_t n = q >> 3;                        // 0..255
      uint32_t c = (q & 7u) * 8u;                 // 0..56 halves
      const void* g = (const void*)(Wt + (uint64_t)(n0 + n) * K_TOTAL + kk + c);
      uint32_t lds_off = (uint32_t)(uintptr_t)(&sB[buf][n * LDB + c]);
      async_copy_b128(lds_off, g);
    }
    // A tile: fp32 load + convert + ds_store (needs the VALU for cvt anyway).
#pragma unroll
    for (int p = 0; p < 8; ++p) {
      uint32_t f4 = tid + (uint32_t)p * 256u;     // 0..2047
      uint32_t r  = f4 >> 4;                      // 0..127
      uint32_t c  = (f4 & 15u) * 4u;              // 0..60
      float4 xv = *reinterpret_cast<const float4*>(
          X + (uint64_t)(m0 + r) * K_TOTAL + kk + c);
      unsigned short* d = &sA[buf][r * LDA + c];
      d[0] = f2bf(xv.x); d[1] = f2bf(xv.y); d[2] = f2bf(xv.z); d[3] = f2bf(xv.w);
    }
  };

  stage(0, 0);
  wait_async_zero();
  __syncthreads();

  int p = 0;
  for (uint32_t kk = 0; kk < K_TOTAL; kk += BK) {
    if (kk + BK < K_TOTAL) stage(kk + BK, p ^ 1);   // fill other buffer

#pragma unroll
    for (uint32_t ks = 0; ks < BK; ks += 32u) {
      // 4 B fragments: lane -> col n = l15; halves = K[lhalf*16 .. +15] contiguous
      union { uint4 q[2]; v16bf v; } bfr[4];
#pragma unroll
      for (int u = 0; u < 4; ++u) {
        uint32_t n  = wn * 64u + (uint32_t)u * 16u + l15;
        uint32_t kb = ks + lhalf * 16u;
        bfr[u].q[0] = *reinterpret_cast<const uint4*>(&sB[p][n * LDB + kb]);
        bfr[u].q[1] = *reinterpret_cast<const uint4*>(&sB[p][n * LDB + kb + 8u]);
      }
#pragma unroll
      for (int t = 0; t < 4; ++t) {
        // A fragment: lane -> row m = l15; halves = K[lhalf*16 .. +15] contiguous
        union { uint4 q[2]; v16bf v; } afr;
        uint32_t m  = wm * 64u + (uint32_t)t * 16u + l15;
        uint32_t kb = ks + lhalf * 16u;
        afr.q[0] = *reinterpret_cast<const uint4*>(&sA[p][m * LDA + kb]);
        afr.q[1] = *reinterpret_cast<const uint4*>(&sA[p][m * LDA + kb + 8u]);
#pragma unroll
        for (int u = 0; u < 4; ++u)
          acc[t][u] = __builtin_amdgcn_wmma_f32_16x16x32_bf16(
              false, afr.v, false, bfr[u].v, (short)0, acc[t][u], false, false);
      }
    }
    wait_async_zero();       // B tile for next iter resident in LDS
    __syncthreads();         // + A-tile dscnt / all waves fenced
    p ^= 1;
  }

  // Epilogue: D layout: VGPR v -> row 8*lhalf + v, lane l15 -> col. Add bias, store fp32.
#pragma unroll
  for (int t = 0; t < 4; ++t)
#pragma unroll
    for (int u = 0; u < 4; ++u) {
      uint32_t col = n0 + wn * 64u + (uint32_t)u * 16u + l15;
      float b = bias[col];
#pragma unroll
      for (int v = 0; v < 8; ++v) {
        uint32_t row = m0 + wm * 64u + (uint32_t)t * 16u + lhalf * 8u + (uint32_t)v;
        Out[(uint64_t)row * N_TOTAL + col] = acc[t][u][v] + b;
      }
    }
}

extern "C" void kernel_launch(void* const* d_in, const int* in_sizes, int n_in,
                              void* d_out, int out_size, void* d_ws, size_t ws_size,
                              hipStream_t stream) {
  const float* x    = (const float*)d_in[0];
  const float* L0   = (const float*)d_in[1];
  const float* R0   = (const float*)d_in[2];
  const float* L1   = (const float*)d_in[3];
  const float* R1   = (const float*)d_in[4];
  const float* L2   = (const float*)d_in[5];
  const float* R2   = (const float*)d_in[6];
  const float* bias = (const float*)d_in[7];
  unsigned short* Wt = (unsigned short*)d_ws;   // 32 MB bf16, transposed [N][K]
  float* out = (float*)d_out;

  (void)in_sizes; (void)n_in; (void)out_size; (void)ws_size;

  KLinear_build_wt<<<(N_TOTAL * K_TOTAL) / 256, 256, 0, stream>>>(L0, R0, L1, R1, L2, R2, Wt);

  dim3 grid(N_TOTAL / BN, M_TOTAL / BM);   // 16 x 128 blocks
  KLinear_gemm<<<grid, 256, 0, stream>>>(x, Wt, bias, out);
}
